// UnifiedTransformerVM_62380105007510
// MI455X (gfx1250) — matrix operand order repaired and checked by reading, not compiled, not run on hardware
//
#include <hip/hip_runtime.h>
#include <hip/hip_bf16.h>

typedef _Float16 h16;
typedef __attribute__((ext_vector_type(16))) _Float16 v16h;
typedef __attribute__((ext_vector_type(8)))  float    v8f;

namespace {
constexpr int kB = 16, kT = 128, kR = 2048, kAB = 20, kVB = 32, kE = 128, kKP = 32;
constexpr float kTemp = 10.0f;
}

// One block = one (batch, 128-read-row) tile. 8 waves, each owns 16 rows.
// Wave 0 replays the sequential write phase (exact-match dedup) into LDS;
// then all waves do scores -> softmax -> AV with v_wmma_f32_16x16x32_f16.
__global__ __launch_bounds__(256) void attn_mem_fused(
    const float* __restrict__ wae,   // [B,T,20] write addr enc (+/-1)
    const float* __restrict__ wve,   // [B,T,32] write val enc (+/-1)
    const float* __restrict__ rae,   // [B,R,20] read addr enc (+/-1)
    float* __restrict__ out)         // [B,R,32]
{
  __shared__ h16 s_keys[kE * kKP];        // keys, K padded 20->32 with 0
  __shared__ h16 s_vals[kE * kKP];        // values (+/-1 exact in f16)
  __shared__ h16 s_ra[128 * kKP];         // this block's read rows, padded
  __shared__ h16 s_w[8][16 * kE];         // per-wave softmax weights [m][e]
  __shared__ unsigned s_addr[kE];
  __shared__ int s_count;

  const int b    = blockIdx.x >> 4;
  const int rb   = blockIdx.x & 15;
  const int tid  = threadIdx.x;
  const int lane = tid & 31;
  const int wid  = tid >> 5;
  const int nlo  = lane & 15;
  const int hi   = lane >> 4;             // which 16-lane half

  // Stage 128 read-address rows: f32 -> f16, pad K 20->32.
  {
    const float* gra = rae + ((size_t)b * kR + (size_t)rb * 128) * kAB;
    for (int i = tid; i < 128 * kKP; i += 256) {
      const int row = i >> 5, col = i & 31;
      s_ra[i] = (col < kAB) ? (h16)gra[row * kAB + col] : (h16)0.f;
    }
  }

  // ---- Sequential write phase (wave 0 only; count <= 128) ----
  if (wid == 0) {
    for (int i = lane; i < kE * kKP; i += 32) { s_keys[i] = (h16)0.f; s_vals[i] = (h16)0.f; }
    int count = 0;
    for (int t = 0; t < kT; ++t) {
      const float* ap = wae + ((size_t)b * kT + t) * kAB;
      const float av = (lane < kAB) ? ap[lane] : -1.0f;
      // decode 20-bit address: score > 19 in the reference <=> exact match
      unsigned bits = (lane < kAB && av > 0.f) ? (1u << lane) : 0u;
      #pragma unroll
      for (int m = 1; m < 32; m <<= 1) bits |= __shfl_xor(bits, m, 32);
      // linear search stored addresses (keys are unique by construction)
      int best = 0x7fffffff;
      for (int j = lane; j < count; j += 32)
        if (s_addr[j] == bits && j < best) best = j;
      #pragma unroll
      for (int m = 1; m < 32; m <<= 1) { const int o = __shfl_xor(best, m, 32); if (o < best) best = o; }
      const bool over = (best != 0x7fffffff);
      const int idx = over ? best : count;
      const float* vp = wve + ((size_t)b * kT + t) * kVB;
      s_vals[idx * kKP + lane] = (h16)vp[lane];       // overwrite-or-append value
      if (!over) {                                    // append key
        s_keys[count * kKP + lane] = (lane < kAB) ? (h16)av : (h16)0.f;
        if (lane == 0) s_addr[count] = bits;
        ++count;
      }
    }
    if (lane == 0) s_count = count;
  }
  __syncthreads();
  const int count = s_count;

  // ---- Scores: 16 rows x 128 entries, 8x v_wmma_f32_16x16x32_f16 (exact) ----
  // A layout (16x32 f16): M=nlo, K = kb + j for j in 0..15, kb = 16*hi
  v16h A;
  {
    const int kb = hi ? 16 : 0;
    const h16* rp = &s_ra[(wid * 16 + nlo) * kKP];
    #pragma unroll
    for (int j = 0; j < 16; ++j) A[j] = rp[kb + j];
  }
  float sc[8][8];                                     // [entry-tile][D-vgpr r]
  #pragma unroll
  for (int t = 0; t < 8; ++t) {
    // B layout (32x16 f16): N = nlo, K half selected by hi; B[k][n] = keys[n][k]
    v16h Bm;
    const h16* cp = &s_keys[(t * 16 + nlo) * kKP + (hi ? 16 : 0)];
    #pragma unroll
    for (int j = 0; j < 16; ++j) Bm[j] = cp[j];
    v8f c = {};
    c = __builtin_amdgcn_wmma_f32_16x16x32_f16(false, A, false, Bm, (short)0, c, false, false);
    const bool valid = (t * 16 + nlo) < count;        // mask entries >= count
    #pragma unroll
    for (int r = 0; r < 8; ++r) sc[t][r] = valid ? c[r] * kTemp : -3.0e38f;
  }

  // ---- Softmax over entries; D element r on this lane is row m = r + 8*hi,
  //      so reductions stay inside the 16-lane half (xor masks 1,2,4,8). ----
  #pragma unroll
  for (int r = 0; r < 8; ++r) {
    float mx = sc[0][r];
    #pragma unroll
    for (int t = 1; t < 8; ++t) mx = fmaxf(mx, sc[t][r]);
    #pragma unroll
    for (int m = 1; m <= 8; m <<= 1) mx = fmaxf(mx, __shfl_xor(mx, m, 32));
    float sum = 0.f;
    #pragma unroll
    for (int t = 0; t < 8; ++t) { const float e = __expf(sc[t][r] - mx); sc[t][r] = e; sum += e; }
    #pragma unroll
    for (int m = 1; m <= 8; m <<= 1) sum += __shfl_xor(sum, m, 32);
    const float inv = 1.0f / sum;
    #pragma unroll
    for (int t = 0; t < 8; ++t) sc[t][r] *= inv;
  }

  // Stage weights to LDS row-major [m][e] so they can re-enter the A layout.
  #pragma unroll
  for (int t = 0; t < 8; ++t)
    #pragma unroll
    for (int r = 0; r < 8; ++r)
      s_w[wid][(r + 8 * hi) * kE + t * 16 + nlo] = (h16)sc[t][r];
  __syncthreads();

  // ---- AV: weights[16x128] @ values[128x32], 8x WMMA (2 N-tiles x 4 K-chunks) ----
  float* ob = out + ((size_t)b * kR + (size_t)rb * 128 + (size_t)wid * 16) * kVB;
  #pragma unroll
  for (int vt = 0; vt < 2; ++vt) {
    v8f acc = {};
    #pragma unroll
    for (int kt = 0; kt < 4; ++kt) {
      v16h Aw;
      const h16* wr = &s_w[wid][nlo * kE + kt * 32 + (hi ? 16 : 0)];
      #pragma unroll
      for (int j = 0; j < 16; ++j) Aw[j] = wr[j];
      v16h Bv;
      const int kb2 = kt * 32 + (hi ? 16 : 0);
      #pragma unroll
      for (int j = 0; j < 16; ++j) Bv[j] = s_vals[(kb2 + j) * kKP + vt * 16 + nlo];
      acc = __builtin_amdgcn_wmma_f32_16x16x32_f16(false, Aw, false, Bv, (short)0, acc, false, false);
    }
    // D layout: vgpr r -> row m = r + 8*hi, col = vt*16 + nlo
    #pragma unroll
    for (int r = 0; r < 8; ++r)
      ob[(r + 8 * hi) * kVB + vt * 16 + nlo] = acc[r];
  }
}

extern "C" void kernel_launch(void* const* d_in, const int* in_sizes, int n_in,
                              void* d_out, int out_size, void* d_ws, size_t ws_size,
                              hipStream_t stream) {
  (void)in_sizes; (void)n_in; (void)d_ws; (void)ws_size; (void)out_size;
  const float* wae = (const float*)d_in[0];   // write_addr_enc [16,128,20]
  const float* wve = (const float*)d_in[1];   // write_val_enc  [16,128,32]
  const float* rae = (const float*)d_in[2];   // read_addr_enc  [16,2048,20]
  attn_mem_fused<<<kB * (kR / 128), 256, 0, stream>>>(wae, wve, rae, (float*)d_out);
}
